// GroupedQueryAttention_53661321396805
// MI455X (gfx1250) — matrix operand order repaired and checked by reading, not compiled, not run on hardware
//
#include <hip/hip_runtime.h>

typedef __attribute__((ext_vector_type(16))) _Float16 v16h;
typedef __attribute__((ext_vector_type(8)))  _Float16 v8h;
typedef __attribute__((ext_vector_type(8)))  float    v8f;
typedef __attribute__((ext_vector_type(4)))  float    f32x4;
typedef __attribute__((ext_vector_type(4)))  unsigned int u32x4;
typedef __attribute__((ext_vector_type(8)))  int      i32x8;
typedef __attribute__((ext_vector_type(4)))  int      i32x4;

#define B_  4
#define S_  2048
#define D_  1024
#define H_  16
#define HK_ 4
#define HD_ 64
#define G_  4

static __device__ __forceinline__ v16h cat8(v8h lo, v8h hi) {
  return __builtin_shufflevector(lo, hi, 0,1,2,3,4,5,6,7,8,9,10,11,12,13,14,15);
}

// ---------------------------------------------------------------------------
// TDM: issue a 2-D f16 tile load (Global -> LDS) via the Tensor Data Mover.
// Descriptor per CDNA5 ISA section 8.3/8.4. 2-D tensor: [d1 rows][d0 cols],
// row stride stride_d0 (elements). Tile [tile_d1 rows][tile_d0 cols] starting
// at gsrc. Optional LDS padding: after every 2^(pad_interval+1) DWORDs insert
// (pad_amount+1) DWORDs.
// Uses the 6-arg builtin form (upstream clang-23 / therock-10.0 headers).
// ---------------------------------------------------------------------------
static __device__ __forceinline__ void tdm_load_tile_f16(
    const void* gsrc, unsigned lds_off_bytes,
    unsigned tensor_d0, unsigned tensor_d1, unsigned stride_d0,
    unsigned tile_d0, unsigned tile_d1,
    unsigned pad_enable, unsigned pad_interval, unsigned pad_amount)
{
  unsigned long long ga = (unsigned long long)(uintptr_t)gsrc;
  u32x4 g0;
  g0[0] = 1u;                                              // count=1, user D#
  g0[1] = lds_off_bytes;                                   // lds_addr
  g0[2] = (unsigned)(ga & 0xFFFFFFFFu);                    // global_addr[31:0]
  g0[3] = (unsigned)((ga >> 32) & 0x01FFFFFFu) | (2u << 30); // addr[56:32] | type=2
  i32x8 g1;
  g1[0] = (int)((1u << 16) |                               // data_size=1 (2 bytes)
                (pad_enable << 20) | (pad_interval << 22) | (pad_amount << 25));
  g1[1] = (int)((tensor_d0 & 0xFFFFu) << 16);              // dim0[15:0]
  g1[2] = (int)(((tensor_d0 >> 16) & 0xFFFFu) | ((tensor_d1 & 0xFFFFu) << 16));
  g1[3] = (int)(((tensor_d1 >> 16) & 0xFFFFu) | ((tile_d0 & 0xFFFFu) << 16));
  g1[4] = (int)(tile_d1 & 0xFFFFu);                        // tile_dim1 (tile_dim2=0)
  g1[5] = (int)stride_d0;                                  // dim0_stride[31:0]
  g1[6] = 0;
  g1[7] = 0;
  i32x4 z4 = {};
  i32x8 z8 = {};
  __builtin_amdgcn_tensor_load_to_lds(g0, g1, z4, z4, z8, 0);
}

// ---------------------------------------------------------------------------
// Generic WMMA GEMM: C[M,N] = A[M,K] @ W[K,N]
// ---------------------------------------------------------------------------
template<bool A_F16, bool OUT_F32>
__global__ __launch_bounds__(256)
void gemm_wmma(const void* __restrict__ Av, const float* __restrict__ W,
               void* __restrict__ Cv, int M, int N, int K)
{
  __shared__ _Float16 lds_a[128][40];   // [m][k], stride 80B (16B-aligned frags)
  __shared__ _Float16 lds_b[64][40];    // [n][k]

  const int tid  = threadIdx.x;
  const int lane = tid & 31;
  const int wave = tid >> 5;        // 0..7, each owns 16 rows x 64 cols
  const int hi16 = lane >> 4;       // 0/1
  const int ln   = lane & 15;
  const int akb  = hi16 * 8;        // A-frag K base within lane (0 or 8)
  const int bkb  = hi16 * 16;       // unused: B frags use akb, same k order as A

  const int m0 = blockIdx.y * 128;
  const int n0 = blockIdx.x * 64;

  v8f acc[4] = {};

  for (int k0 = 0; k0 < K; k0 += 32) {
    // ---- stage A tile (128 x 32) ----
    {
      const int row  = tid >> 1;
      const int hseg = (tid & 1) * 16;
      if constexpr (A_F16) {
        const _Float16* src = (const _Float16*)Av + (size_t)(m0 + row) * K + k0 + hseg;
        *(v8h*)&lds_a[row][hseg]     = *(const v8h*)(src);
        *(v8h*)&lds_a[row][hseg + 8] = *(const v8h*)(src + 8);
        if (k0 + 32 < K) __builtin_prefetch(src + 32, 0, 1);
      } else {
        const float* src = (const float*)Av + (size_t)(m0 + row) * K + k0 + hseg;
        #pragma unroll
        for (int c = 0; c < 16; c += 4) {
          f32x4 v = *(const f32x4*)(src + c);
          lds_a[row][hseg + c + 0] = (_Float16)v[0];
          lds_a[row][hseg + c + 1] = (_Float16)v[1];
          lds_a[row][hseg + c + 2] = (_Float16)v[2];
          lds_a[row][hseg + c + 3] = (_Float16)v[3];
        }
        if (k0 + 32 < K) __builtin_prefetch(src + 32, 0, 1);
      }
    }
    // ---- stage W tile (32 x 64) transposed into lds_b[n][k] ----
    {
      const int kk = tid & 31;
      const int ng = (tid >> 5) * 8;
      const float* src = W + (size_t)(k0 + kk) * N + n0 + ng;
      f32x4 w0 = *(const f32x4*)(src);
      f32x4 w1 = *(const f32x4*)(src + 4);
      #pragma unroll
      for (int j = 0; j < 4; ++j) {
        lds_b[ng + j][kk]     = (_Float16)w0[j];
        lds_b[ng + 4 + j][kk] = (_Float16)w1[j];
      }
      if (k0 + 32 < K) __builtin_prefetch(src + (size_t)32 * N, 0, 1);
    }
    __syncthreads();

    // ---- fragments + WMMA ----
    const int arow = wave * 16 + ln;
    v8h alo = *(const v8h*)&lds_a[arow][akb];
    v8h ahi = *(const v8h*)&lds_a[arow][akb + 16];
    v16h afrag = cat8(alo, ahi);

    #pragma unroll
    for (int j = 0; j < 4; ++j) {
      const int bcol = j * 16 + ln;
      v8h blo = *(const v8h*)&lds_b[bcol][akb];
      v8h bhi = *(const v8h*)&lds_b[bcol][akb + 16];
      v16h bfrag = cat8(blo, bhi);
      acc[j] = __builtin_amdgcn_wmma_f32_16x16x32_f16(
          false, afrag, false, bfrag, (short)0, acc[j], false, false);
    }
    __syncthreads();
  }

  // ---- epilogue ----
  #pragma unroll
  for (int j = 0; j < 4; ++j) {
    #pragma unroll
    for (int i = 0; i < 8; ++i) {
      const int row = m0 + wave * 16 + i + hi16 * 8;
      const int col = n0 + j * 16 + ln;
      if constexpr (OUT_F32)
        ((float*)Cv)[(size_t)row * N + col] = acc[j][i];
      else
        ((_Float16*)Cv)[(size_t)row * N + col] = (_Float16)acc[j][i];
    }
  }
}

// ---------------------------------------------------------------------------
// GQA flash attention.
// Grid: (S/64, H, B). Block: 128 threads (4 waves). Each wave: 16 q-rows.
// KV block of 32 keys per iteration. K block staged by the Tensor Data Mover
// (with TDM padding producing the 72-halfword LDS row stride); V staged
// manually transposed. All matmuls via v_wmma_f32_16x16x32_f16.
// ---------------------------------------------------------------------------
__global__ __launch_bounds__(128)
void attn_wmma(const _Float16* __restrict__ Qp, const _Float16* __restrict__ Kp,
               const _Float16* __restrict__ Vp, _Float16* __restrict__ ctx)
{
  __shared__ _Float16 lds_k[32][72];       // [key][hd]   (B-operand for Q@K^T)
  __shared__ _Float16 lds_v[64][40];       // [hd][key]   (B-operand for P@V)
  __shared__ _Float16 lds_p[4][16][40];    // per-wave P relayout buffer

  const int tid  = threadIdx.x;
  const int lane = tid & 31;
  const int wave = tid >> 5;        // 0..3
  const int hi16 = lane >> 4;
  const int ln   = lane & 15;
  const int akb  = hi16 * 8;
  const int bkb  = hi16 * 16;

  const int h   = blockIdx.y;
  const int b   = blockIdx.z;
  const int kvh = h / G_;
  const int q0  = blockIdx.x * 64;

  const unsigned lds_k_off = (unsigned)(uintptr_t)(&lds_k[0][0]);

  // ---- load Q fragments once (16 rows x 64 hd per wave, scale folded in) ----
  const int qrow = q0 + wave * 16 + ln;
  const _Float16* qptr = Qp + ((size_t)(b * S_ + qrow)) * D_ + h * HD_;
  v8h qa_lo = *(const v8h*)(qptr + akb);
  v8h qa_hi = *(const v8h*)(qptr + akb + 16);
  v8h qb_lo = *(const v8h*)(qptr + 32 + akb);
  v8h qb_hi = *(const v8h*)(qptr + 48 + akb);
  v16h qfa = cat8(qa_lo, qa_hi);
  v16h qfb = cat8(qb_lo, qb_hi);
  const _Float16 qs = (_Float16)0.125f;    // 1/sqrt(64)
  #pragma unroll
  for (int i = 0; i < 16; ++i) { qfa[i] *= qs; qfb[i] *= qs; }

  float mrow[8], lrow[8];
  v8f o[4] = {};
  #pragma unroll
  for (int i = 0; i < 8; ++i) { mrow[i] = -1e30f; lrow[i] = 0.0f; }

  for (int k0 = 0; k0 < S_; k0 += 32) {
    // ---- stage K block [key][hd] via Tensor Data Mover (wave 0 issues) ----
    if (tid < 32) {
      const _Float16* ksrc = Kp + ((size_t)(b * S_ + k0)) * (HK_ * HD_) + kvh * HD_;
      // tensor: [B*S rows][256 cols] f16, row stride 256 elements.
      // tile: 32 rows x 64 cols. LDS pad: every 32 DWORDs (+4 DWORDs) -> 72-h stride.
      tdm_load_tile_f16(ksrc, lds_k_off,
                        /*tensor_d0=*/HK_ * HD_, /*tensor_d1=*/B_ * S_,
                        /*stride_d0=*/HK_ * HD_,
                        /*tile_d0=*/HD_, /*tile_d1=*/32,
                        /*pad_enable=*/1, /*pad_interval=*/4, /*pad_amount=*/3);
    }
    // ---- stage V block transposed [hd][key] ----
    {
      const int key = tid & 31;
      const int hd0 = (tid >> 5) * 16;
      const _Float16* src = Vp + ((size_t)(b * S_ + k0 + key)) * (HK_ * HD_) + kvh * HD_ + hd0;
      v8h v0 = *(const v8h*)(src);
      v8h v1 = *(const v8h*)(src + 8);
      #pragma unroll
      for (int j = 0; j < 8; ++j) {
        lds_v[hd0 + j][key]     = v0[j];
        lds_v[hd0 + 8 + j][key] = v1[j];
      }
    }
    if (tid < 32) __builtin_amdgcn_s_wait_tensorcnt(0);
    __syncthreads();

    // ---- scores S = (Q*scale) @ K^T : 16 x 32 per wave ----
    v8f sc[2];
    #pragma unroll
    for (int c = 0; c < 2; ++c) {
      const int keyc = c * 16 + ln;
      v8h ka_lo = *(const v8h*)&lds_k[keyc][akb];
      v8h ka_hi = *(const v8h*)&lds_k[keyc][akb + 16];
      v8h kb_lo = *(const v8h*)&lds_k[keyc][32 + akb];
      v8h kb_hi = *(const v8h*)&lds_k[keyc][32 + akb + 16];
      v16h kfa = cat8(ka_lo, ka_hi);
      v16h kfb = cat8(kb_lo, kb_hi);
      v8f z = {};
      z = __builtin_amdgcn_wmma_f32_16x16x32_f16(false, qfa, false, kfa, (short)0, z, false, false);
      z = __builtin_amdgcn_wmma_f32_16x16x32_f16(false, qfb, false, kfb, (short)0, z, false, false);
      sc[c] = z;
    }

    // ---- online softmax (row reductions across 16-lane halves) ----
    float pr0[8], pr1[8];
    #pragma unroll
    for (int i = 0; i < 8; ++i) {
      float mi = fmaxf(sc[0][i], sc[1][i]);
      mi = fmaxf(mi, __shfl_xor(mi, 1, 32));
      mi = fmaxf(mi, __shfl_xor(mi, 2, 32));
      mi = fmaxf(mi, __shfl_xor(mi, 4, 32));
      mi = fmaxf(mi, __shfl_xor(mi, 8, 32));
      const float nm    = fmaxf(mrow[i], mi);
      const float alpha = __expf(mrow[i] - nm);
      mrow[i] = nm;
      const float p0 = __expf(sc[0][i] - nm);
      const float p1 = __expf(sc[1][i] - nm);
      pr0[i] = p0; pr1[i] = p1;
      float rs = p0 + p1;
      rs += __shfl_xor(rs, 1, 32);
      rs += __shfl_xor(rs, 2, 32);
      rs += __shfl_xor(rs, 4, 32);
      rs += __shfl_xor(rs, 8, 32);
      lrow[i] = lrow[i] * alpha + rs;
      #pragma unroll
      for (int j = 0; j < 4; ++j) o[j][i] *= alpha;
    }

    // ---- P: C-layout -> LDS -> A-operand layout (per-wave region) ----
    #pragma unroll
    for (int i = 0; i < 8; ++i) {
      lds_p[wave][i + hi16 * 8][ln]      = (_Float16)pr0[i];
      lds_p[wave][i + hi16 * 8][16 + ln] = (_Float16)pr1[i];
    }
    v8h plo = *(const v8h*)&lds_p[wave][ln][akb];
    v8h phi = *(const v8h*)&lds_p[wave][ln][akb + 16];
    v16h pfrag = cat8(plo, phi);

    // ---- O += P @ V ----
    #pragma unroll
    for (int j = 0; j < 4; ++j) {
      const int hdc = j * 16 + ln;
      v8h vlo = *(const v8h*)&lds_v[hdc][akb];
      v8h vhi = *(const v8h*)&lds_v[hdc][akb + 16];
      v16h vfrag = cat8(vlo, vhi);
      o[j] = __builtin_amdgcn_wmma_f32_16x16x32_f16(false, pfrag, false, vfrag, (short)0, o[j], false, false);
    }
    __syncthreads();
  }

  // ---- normalize + store context (f16) ----
  float inv[8];
  #pragma unroll
  for (int i = 0; i < 8; ++i) inv[i] = 1.0f / lrow[i];
  #pragma unroll
  for (int j = 0; j < 4; ++j) {
    #pragma unroll
    for (int i = 0; i < 8; ++i) {
      const int q   = q0 + wave * 16 + i + hi16 * 8;
      const int col = h * HD_ + j * 16 + ln;
      ctx[((size_t)(b * S_ + q)) * D_ + col] = (_Float16)(o[j][i] * inv[i]);
    }
  }
}

// ---------------------------------------------------------------------------
extern "C" void kernel_launch(void* const* d_in, const int* in_sizes, int n_in,
                              void* d_out, int out_size, void* d_ws, size_t ws_size,
                              hipStream_t stream)
{
  const float* query = (const float*)d_in[0];
  const float* key   = (const float*)d_in[1];
  const float* value = (const float*)d_in[2];
  const float* Wq    = (const float*)d_in[3];
  const float* Wk    = (const float*)d_in[4];
  const float* Wv    = (const float*)d_in[5];
  const float* Wo    = (const float*)d_in[6];
  float* out = (float*)d_out;

  char* ws = (char*)d_ws;
  _Float16* Qp  = (_Float16*)(ws);                              // 16 MB
  _Float16* Kp  = (_Float16*)(ws + (size_t)16 * 1024 * 1024);   //  4 MB
  _Float16* Vp  = (_Float16*)(ws + (size_t)20 * 1024 * 1024);   //  4 MB
  _Float16* ctx = (_Float16*)(ws + (size_t)24 * 1024 * 1024);   // 16 MB

  const int M = B_ * S_;           // 8192
  const int Nkv = HK_ * HD_;       // 256

  gemm_wmma<false, false><<<dim3(D_ / 64, M / 128), 256, 0, stream>>>(query, Wq, Qp, M, D_, D_);
  gemm_wmma<false, false><<<dim3(Nkv / 64, M / 128), 256, 0, stream>>>(key,   Wk, Kp, M, Nkv, D_);
  gemm_wmma<false, false><<<dim3(Nkv / 64, M / 128), 256, 0, stream>>>(value, Wv, Vp, M, Nkv, D_);

  attn_wmma<<<dim3(S_ / 64, H_, B_), 128, 0, stream>>>(Qp, Kp, Vp, ctx);

  gemm_wmma<true, true><<<dim3(D_ / 64, M / 128), 256, 0, stream>>>(ctx, Wo, out, M, D_, D_);
}
